// RelationUnit_43404939493428
// MI455X (gfx1250) — matrix operand order repaired and checked by reading, not compiled, not run on hardware
//
#include <hip/hip_runtime.h>
#include <math.h>

typedef __attribute__((ext_vector_type(16))) _Float16 v16h;
typedef __attribute__((ext_vector_type(8)))  _Float16 v8h;
typedef __attribute__((ext_vector_type(8)))  float    v8f;
typedef __attribute__((ext_vector_type(4)))  float    v4f;

// B=8, K=512, D_APP=768, D_K=96, D_G=96
// ws layout (bytes):
//   [0)            w_k  f16  4096*96*2 = 786432
//   [786432)       w_q  f16             786432
//   [1572864)      w_v  f32  4096*96*4 = 1572864
//   [3145728)      partials: 1024 records * 1568 floats (M[16],Z[16],acc[16][96])
// total ~9.6 MB

// ---------------------------------------------------------------------------
// Kernel 1: projections  out[r,c] = f_a[r,:] . W[c,:] + b[c]   (r<4096, c<96)
// One wave per (rowtile, matrix); A fragment reused across all 6 col tiles.
// ---------------------------------------------------------------------------
__global__ __launch_bounds__(128)
void k_proj(const float* __restrict__ fa,
            const float* __restrict__ Wk, const float* __restrict__ bk,
            const float* __restrict__ Wq, const float* __restrict__ bq,
            const float* __restrict__ Wv, const float* __restrict__ bv,
            _Float16* __restrict__ wk_h, _Float16* __restrict__ wq_h,
            float* __restrict__ wv_f)
{
    int w    = blockIdx.x * 4 + (threadIdx.x >> 5);   // 0..767
    int lane = threadIdx.x & 31;
    int mat     = w / 256;                            // 0=K,1=Q,2=V
    int rowtile = w % 256;
    const float* W;  const float* bias;
    if      (mat == 0) { W = Wk; bias = bk; }
    else if (mat == 1) { W = Wq; bias = bq; }
    else               { W = Wv; bias = bv; }

    int half = lane >> 4;
    int nn   = lane & 15;
    int r0   = rowtile * 16;
    int row  = r0 + nn;                               // A row = lane%16

    v8f acc[6];
    #pragma unroll
    for (int c = 0; c < 6; ++c) acc[c] = {};

    for (int kk = 0; kk < 768; kk += 32) {
        // A fragment (16x32 f16): elems 0..15 -> K = kk+16h+0..15
        const float* ap = fa + (size_t)row * 768 + kk + 16 * half;
        v4f a0 = *(const v4f*)(ap);
        v4f a1 = *(const v4f*)(ap + 4);
        v4f a2 = *(const v4f*)(ap + 8);
        v4f a3 = *(const v4f*)(ap + 12);
        v16h af;
        #pragma unroll
        for (int e = 0; e < 4; ++e) {
            af[e]      = (_Float16)a0[e];
            af[4 + e]  = (_Float16)a1[e];
            af[8 + e]  = (_Float16)a2[e];
            af[12 + e] = (_Float16)a3[e];
        }
        #pragma unroll
        for (int ct = 0; ct < 6; ++ct) {
            int c = ct * 16 + nn;                     // B col = lane%16
            const float* wp = W + (size_t)c * 768 + kk + 16 * half;
            v16h bf;
            #pragma unroll
            for (int e4 = 0; e4 < 4; ++e4) {
                v4f t = *(const v4f*)(wp + 4 * e4);
                bf[4 * e4 + 0] = (_Float16)t[0];
                bf[4 * e4 + 1] = (_Float16)t[1];
                bf[4 * e4 + 2] = (_Float16)t[2];
                bf[4 * e4 + 3] = (_Float16)t[3];
            }
            acc[ct] = __builtin_amdgcn_wmma_f32_16x16x32_f16(
                false, af, false, bf, (short)0, acc[ct], false, false);
        }
    }
    // D layout: col = lane%16, row = r + 8*half
    #pragma unroll
    for (int ct = 0; ct < 6; ++ct) {
        int c = ct * 16 + nn;
        float bb = bias[c];
        #pragma unroll
        for (int r = 0; r < 8; ++r) {
            int rr = r0 + r + 8 * half;
            float v = acc[ct][r] + bb;
            size_t idx = (size_t)rr * 96 + c;
            if      (mat == 2) wv_f[idx] = v;
            else if (mat == 0) wk_h[idx] = (_Float16)v;
            else               wq_h[idx] = (_Float16)v;
        }
    }
}

// ---------------------------------------------------------------------------
// Kernel 2: fused gate + QK^T (WMMA) + online softmax over m + P^T V partial.
// One wave per (b, 16-col n-tile, m-quarter).  1024 waves total.
// ---------------------------------------------------------------------------
__global__ __launch_bounds__(128)
void k_attn(const float* __restrict__ pe,  const float* __restrict__ wgw,
            const float* __restrict__ wgb,
            const _Float16* __restrict__ wk_h, const _Float16* __restrict__ wq_h,
            const float* __restrict__ wv_f, float* __restrict__ part)
{
    __shared__ float s_wg[96];
    if (threadIdx.x < 96) s_wg[threadIdx.x] = wgw[threadIdx.x];
    __syncthreads();

    int w    = blockIdx.x * 4 + (threadIdx.x >> 5);   // 0..1023
    int lane = threadIdx.x & 31;
    int b     = w >> 7;
    int rem   = w & 127;
    int ntile = rem >> 2;                             // 0..31
    int mq    = rem & 3;                              // m-quarter
    int half  = lane >> 4;
    int nn    = lane & 15;
    int n     = ntile * 16 + nn;
    int rowbase = b * 512;
    int d0    = 48 * half;
    float gb  = wgb[0];
    const float INVS = 0.1020620726159658f;           // 1/sqrt(96)
    const size_t PF = (size_t)16 * 512 * 96;          // next m-chunk stride (floats)

    // Q (B) fragments: col = lane%16, K run = 32t + 16h .. +15 over d
    v16h bqf[3];
    #pragma unroll
    for (int t = 0; t < 3; ++t) {
        const _Float16* qp = wq_h + (size_t)(rowbase + n) * 96 + 32 * t + 16 * half;
        v8h q0 = *(const v8h*)(qp);
        v8h q1 = *(const v8h*)(qp + 8);
        #pragma unroll
        for (int e = 0; e < 8; ++e) { bqf[t][e] = q0[e]; bqf[t][8 + e] = q1[e]; }
    }

    float Mrun = -INFINITY, Z = 0.0f;
    v4f acc[12];
    #pragma unroll
    for (int j = 0; j < 12; ++j) acc[j] = {};

    int mbase = mq * 128;
    for (int chunk = 0; chunk < 8; ++chunk) {
        int m0 = mbase + chunk * 16;
        // S = w_k[m-tile] @ w_q[n-tile]^T over d=96 (3 WMMAs)
        v8f S = {};
        #pragma unroll
        for (int t = 0; t < 3; ++t) {
            const _Float16* kp = wk_h + (size_t)(rowbase + m0 + nn) * 96 + 32 * t + 16 * half;
            v8h a0 = *(const v8h*)(kp);
            v8h a1 = *(const v8h*)(kp + 8);
            v16h af;
            #pragma unroll
            for (int e = 0; e < 8; ++e) { af[e] = a0[e]; af[8 + e] = a1[e]; }
            S = __builtin_amdgcn_wmma_f32_16x16x32_f16(
                false, af, false, bqf[t], (short)0, S, false, false);
        }
        // gate dot (the 805MB stream) + logits; lane element r -> (m0+r+8h, n)
        float l[8];
        #pragma unroll
        for (int r = 0; r < 8; ++r) {
            int mrow = m0 + r + 8 * half;
            const float* prow = pe + (((size_t)(rowbase + mrow)) * 512 + n) * 96;
            if (chunk < 7) {
                // prefetch the 6 cachelines of this lane's row, one m-chunk ahead
                #pragma unroll
                for (int c = 0; c < 6; ++c)
                    __builtin_prefetch(prow + PF + 16 * c, 0, 0);
            }
            float dot = gb;
            #pragma unroll
            for (int g = 0; g < 24; ++g) {
                v4f pv = __builtin_nontemporal_load((const v4f*)(prow + 4 * g));
                dot += pv[0] * s_wg[4 * g + 0] + pv[1] * s_wg[4 * g + 1]
                     + pv[2] * s_wg[4 * g + 2] + pv[3] * s_wg[4 * g + 3];
            }
            // relu then clip(1e-6) then log  == log(max(dot, 1e-6))
            l[r] = __logf(fmaxf(dot, 1e-6f)) + S[r] * INVS;
        }
        // online softmax over m (column n lives in lanes {L, L^16})
        float cmax = l[0];
        #pragma unroll
        for (int r = 1; r < 8; ++r) cmax = fmaxf(cmax, l[r]);
        cmax = fmaxf(cmax, __shfl_xor(cmax, 16, 32));
        float Mnew  = fmaxf(Mrun, cmax);
        float scale = __expf(Mrun - Mnew);
        Z *= scale;
        #pragma unroll
        for (int j = 0; j < 12; ++j) acc[j] *= scale;
        float p[8], psum = 0.0f;
        #pragma unroll
        for (int r = 0; r < 8; ++r) { p[r] = __expf(l[r] - Mnew); psum += p[r]; }
        psum += __shfl_xor(psum, 16, 32);
        Z += psum;
        // acc[n, d0..d0+47] += sum_m p[m,n] * w_v[m, d]
        #pragma unroll
        for (int r = 0; r < 8; ++r) {
            float po = p[r];
            float pp = __shfl_xor(p[r], 16, 32);
            int rowO = m0 + 8 * half + r;
            int rowP = m0 + 8 * (1 - half) + r;
            const float* v0 = wv_f + (size_t)(rowbase + rowO) * 96 + d0;
            const float* v1 = wv_f + (size_t)(rowbase + rowP) * 96 + d0;
            #pragma unroll
            for (int j = 0; j < 12; ++j) {
                v4f x0 = *(const v4f*)(v0 + 4 * j);
                v4f x1 = *(const v4f*)(v1 + 4 * j);
                acc[j] += po * x0 + pp * x1;
            }
        }
        Mrun = Mnew;
    }
    // write partial record: M[16], Z[16], acc[16][96]
    float* pb = part + (size_t)((b * 32 + ntile) * 4 + mq) * 1568;
    if (half == 0) { pb[nn] = Mrun; pb[16 + nn] = Z; }
    #pragma unroll
    for (int j = 0; j < 12; ++j)
        *(v4f*)(pb + 32 + nn * 96 + d0 + 4 * j) = acc[j];
}

// ---------------------------------------------------------------------------
// Kernel 3: merge the four m-quarter partials (log-sum-exp) and normalize.
// ---------------------------------------------------------------------------
__global__ __launch_bounds__(96)
void k_combine(const float* __restrict__ part, float* __restrict__ out)
{
    int bn = blockIdx.x;            // 0..4095 = b*512 + n
    int d  = threadIdx.x;           // 0..95
    int b = bn >> 9, n = bn & 511;
    int ntile = n >> 4, nn = n & 15;
    const float* pq[4];
    #pragma unroll
    for (int q = 0; q < 4; ++q)
        pq[q] = part + (size_t)((b * 32 + ntile) * 4 + q) * 1568;

    float M = -INFINITY;
    #pragma unroll
    for (int q = 0; q < 4; ++q) M = fmaxf(M, pq[q][nn]);
    float Z = 0.0f, a = 0.0f;
    #pragma unroll
    for (int q = 0; q < 4; ++q) {
        float e = __expf(pq[q][nn] - M);
        Z += pq[q][16 + nn] * e;
        a += pq[q][32 + nn * 96 + d] * e;
    }
    out[(size_t)bn * 96 + d] = a / Z;
}

// ---------------------------------------------------------------------------
extern "C" void kernel_launch(void* const* d_in, const int* in_sizes, int n_in,
                              void* d_out, int out_size, void* d_ws, size_t ws_size,
                              hipStream_t stream) {
    const float* fa  = (const float*)d_in[0];
    const float* pe  = (const float*)d_in[1];
    const float* wgw = (const float*)d_in[2];
    const float* wgb = (const float*)d_in[3];
    const float* Wk  = (const float*)d_in[4];
    const float* bk  = (const float*)d_in[5];
    const float* Wq  = (const float*)d_in[6];
    const float* bq  = (const float*)d_in[7];
    const float* Wv  = (const float*)d_in[8];
    const float* bv  = (const float*)d_in[9];

    char* ws = (char*)d_ws;
    _Float16* wk_h = (_Float16*)(ws + 0);
    _Float16* wq_h = (_Float16*)(ws + 786432);
    float*    wv_f = (float*)(ws + 1572864);
    float*    part = (float*)(ws + 3145728);
    float*    out  = (float*)d_out;

    k_proj   <<<192, 128, 0, stream>>>(fa, Wk, bk, Wq, bq, Wv, bv, wk_h, wq_h, wv_f);
    k_attn   <<<256, 128, 0, stream>>>(pe, wgw, wgb, wk_h, wq_h, wv_f, part);
    k_combine<<<4096, 96, 0, stream>>>(part, out);
}
